// Net_62130996904552
// MI455X (gfx1250) — matrix operand order repaired and checked by reading, not compiled, not run on hardware
//
#include <hip/hip_runtime.h>

typedef __attribute__((ext_vector_type(16))) _Float16 v16h;
typedef __attribute__((ext_vector_type(8)))  float    v8f;

#define APPNP_ALPHA 0.1f

// =====================================================================
// WMMA GEMM: C[Mpad,N] = act(A[M,K] @ W[K,N] + bias); optional 2nd output.
// One 16x16 output tile per wave (wave32), V_WMMA_F32_16X16X32_F16,
// K consumed in steps of 32. 256-thread block = 8 waves = 128-row slab.
// A-row index is CLAMPED (branch-free b128 loads); the output buffers
// are padded to a multiple of 128 rows, so the epilogue stores are
// completely unguarded (garbage rows >= M are never read downstream).
// =====================================================================
template<int K, int N, bool RELU, bool DUAL>
__global__ __launch_bounds__(256) void wmma_gemm_kernel(
    const float* __restrict__ A, const float* __restrict__ W,
    const float* __restrict__ bias, float* __restrict__ C,
    float* __restrict__ C2, int M)
{
    const int wave = threadIdx.x >> 5;
    const int lane = threadIdx.x & 31;
    const int half = lane >> 4;          // which 16-lane half of the wave
    const int r    = lane & 15;

    const int rowBase = (blockIdx.x * 8 + wave) * 16;
    const int colBase = blockIdx.y * 16;

    int arow = rowBase + r;
    if (arow >= M) arow = M - 1;         // clamp: branch-free A loads
    const float4* Arow = (const float4*)(A + (size_t)arow * K);
    const float*  Wc   = W + colBase + r;

    v8f acc = {};
    for (int k0 = 0; k0 < K; k0 += 32) {
        // A fragment (16x32 f16): lanes 0-15 (row M=r) hold K = k0+0..15,
        // lanes 16-31 (row M=r) hold K = k0+16..31.
        const int q0 = (k0 + (half << 4)) >> 2;        // first 8-float run
        const int q1 = q0 + 2;                         // second 8-float run
        float4 a0 = Arow[q0];
        float4 a1 = Arow[q0 + 1];
        float4 a2 = Arow[q1];
        float4 a3 = Arow[q1 + 1];

        v16h a;
        a[0]  = (_Float16)a0.x; a[1]  = (_Float16)a0.y;
        a[2]  = (_Float16)a0.z; a[3]  = (_Float16)a0.w;
        a[4]  = (_Float16)a1.x; a[5]  = (_Float16)a1.y;
        a[6]  = (_Float16)a1.z; a[7]  = (_Float16)a1.w;
        a[8]  = (_Float16)a2.x; a[9]  = (_Float16)a2.y;
        a[10] = (_Float16)a2.z; a[11] = (_Float16)a2.w;
        a[12] = (_Float16)a3.x; a[13] = (_Float16)a3.y;
        a[14] = (_Float16)a3.z; a[15] = (_Float16)a3.w;

        // B fragment (32x16 f16): lanes 0-15 (col N=r) hold K=k0+0..15,
        // lanes 16-31 (col N=r) hold K=k0+16..31. Lane-coalesced rows of W.
        v16h b;
#pragma unroll
        for (int e = 0; e < 16; ++e) {
            int k = k0 + e + (half << 4);
            b[e] = (_Float16)Wc[(size_t)k * N];
        }

        acc = __builtin_amdgcn_wmma_f32_16x16x32_f16(
            /*neg_a=*/false, a, /*neg_b=*/false, b,
            /*c_mod=*/(short)0, acc, /*reuse_a=*/false, /*reuse_b=*/false);
    }

    // D layout (32-bit 16x16): VGPR v holds M = v + half*8, N = r.
    const int   n  = colBase + r;
    const float bn = bias[n];                       // hoisted, one load
    float* Cp  = C  + (size_t)(rowBase + (half << 3)) * N + n;
    float* C2p = DUAL ? (C2 + (size_t)(rowBase + (half << 3)) * N + n) : nullptr;
#pragma unroll
    for (int v = 0; v < 8; ++v) {
        float val = acc[v] + bn;
        if (RELU) val = fmaxf(val, 0.0f);
        Cp[(size_t)v * N] = val;                    // unguarded: padded buffer
        if (DUAL) C2p[(size_t)v * N] = val;
    }
}

// ============================ degree / norm ============================
__global__ void deg_init_kernel(float* __restrict__ deg, int n) {
    int i = blockIdx.x * blockDim.x + threadIdx.x;
    if (i < n) deg[i] = 1.0f;                 // self-loop contribution
}

__global__ void deg_scatter_kernel(const int* __restrict__ dst,
                                   float* __restrict__ deg, int E) {
    int e = blockIdx.x * blockDim.x + threadIdx.x;
    if (e < E) atomicAdd(&deg[dst[e]], 1.0f);
}

__global__ void dinv_kernel(const float* __restrict__ deg,
                            float* __restrict__ dinv, int n) {
    int i = blockIdx.x * blockDim.x + threadIdx.x;
    if (i < n) dinv[i] = rsqrtf(deg[i]);
}

// ========================= propagation step ===========================
__global__ void zero_kernel(float* __restrict__ p, int n) {
    int i = blockIdx.x * blockDim.x + threadIdx.x;
    if (i < n) p[i] = 0.0f;
}

// 64 threads per edge: coalesced gather of h[src,:], fp32 atomic scatter.
__global__ __launch_bounds__(256) void edge_scatter_kernel(
    const int* __restrict__ src, const int* __restrict__ dst,
    const float* __restrict__ dinv, const float* __restrict__ h,
    float* __restrict__ agg, int E)
{
    int t = blockIdx.x * blockDim.x + threadIdx.x;
    int e = t >> 6;
    int c = t & 63;
    if (e >= E) return;
    int s = src[e], d = dst[e];
    __builtin_prefetch(&h[(size_t)s * 64], 0, 0);   // global_prefetch_b8
    float w = dinv[s] * dinv[d];
    atomicAdd(&agg[(size_t)d * 64 + c], w * h[(size_t)s * 64 + c]);
}

// out = (1-a)*(agg + dinv^2 * h) + a*h0   (self-loop edge folded in here)
__global__ void combine_kernel(const float* __restrict__ agg,
                               const float* __restrict__ h,
                               const float* __restrict__ h0,
                               const float* __restrict__ dinv,
                               float* __restrict__ out, int n64)
{
    int t = blockIdx.x * blockDim.x + threadIdx.x;
    if (t >= n64) return;
    int i = t >> 6;
    float di = dinv[i];
    out[t] = (1.0f - APPNP_ALPHA) * (agg[t] + di * di * h[t])
             + APPNP_ALPHA * h0[t];
}

// ============================ log-softmax =============================
// One wave (32 lanes) per 64-wide row; shfl_xor reductions (wave32).
__global__ __launch_bounds__(256) void log_softmax_kernel(
    const float* __restrict__ h, float* __restrict__ out, int M)
{
    int wave = threadIdx.x >> 5;
    int lane = threadIdx.x & 31;
    int row  = blockIdx.x * 8 + wave;
    if (row >= M) return;
    const float* p = h + (size_t)row * 64;
    float v0 = p[lane], v1 = p[lane + 32];
    float m = fmaxf(v0, v1);
#pragma unroll
    for (int off = 16; off > 0; off >>= 1) m = fmaxf(m, __shfl_xor(m, off, 32));
    float s = __expf(v0 - m) + __expf(v1 - m);
#pragma unroll
    for (int off = 16; off > 0; off >>= 1) s += __shfl_xor(s, off, 32);
    float lse = m + __logf(s);
    float* q = out + (size_t)row * 64;
    q[lane]      = v0 - lse;
    q[lane + 32] = v1 - lse;
}

// =============================== driver ===============================
extern "C" void kernel_launch(void* const* d_in, const int* in_sizes, int n_in,
                              void* d_out, int out_size, void* d_ws, size_t ws_size,
                              hipStream_t stream)
{
    const float* x  = (const float*)d_in[0];   // [N,512]
    const float* W1 = (const float*)d_in[1];   // [512,256]
    const float* b1 = (const float*)d_in[2];   // [256]
    const float* W2 = (const float*)d_in[3];   // [256,64]
    const float* b2 = (const float*)d_in[4];   // [64]
    const int* edge = (const int*)d_in[5];     // [2,E] flat

    const int N = 100000;
    const int E = in_sizes[5] / 2;
    const int H = 256, C = 64, STEPS = 10;
    const int* src = edge;
    const int* dst = edge + E;

    const int msl  = (N + 127) / 128;   // M slabs of 128 rows
    const int Mpad = msl * 128;         // padded rows -> unguarded GEMM stores

    // workspace carve-out (row-padded where GEMMs write)
    char* ws = (char*)d_ws;
    size_t off = 0;
    auto carve = [&](size_t bytes) -> float* {
        float* p = (float*)(ws + off);
        off += (bytes + 255) & ~(size_t)255;
        return p;
    };
    float* h1   = carve((size_t)Mpad * H * sizeof(float));   // MLP hidden
    float* h0   = carve((size_t)Mpad * C * sizeof(float));   // teleport anchor
    float* hA   = carve((size_t)Mpad * C * sizeof(float));   // ping
    float* hB   = carve((size_t)N    * C * sizeof(float));   // pong
    float* agg  = carve((size_t)N    * C * sizeof(float));   // scatter accum
    float* deg  = carve((size_t)N * sizeof(float));
    float* dinv = carve((size_t)N * sizeof(float));

    // MLP layer 1: h1 = relu(x @ W1 + b1)
    wmma_gemm_kernel<512, 256, true, false><<<dim3(msl, H / 16), 256, 0, stream>>>(
        x, W1, b1, h1, nullptr, N);
    // MLP layer 2: h0 = h1 @ W2 + b2 (also seeds hA = h0)
    wmma_gemm_kernel<256, 64, false, true><<<dim3(msl, C / 16), 256, 0, stream>>>(
        h1, W2, b2, h0, hA, N);

    // GCN normalization: deg (with self-loops) -> dinv
    deg_init_kernel<<<(N + 255) / 256, 256, 0, stream>>>(deg, N);
    deg_scatter_kernel<<<(E + 255) / 256, 256, 0, stream>>>(dst, deg, E);
    dinv_kernel<<<(N + 255) / 256, 256, 0, stream>>>(deg, dinv, N);

    // APPNP propagation
    const int n64 = N * C;
    float* hcur = hA;
    float* hnxt = hB;
    for (int k = 0; k < STEPS; ++k) {
        zero_kernel<<<(n64 + 255) / 256, 256, 0, stream>>>(agg, n64);
        edge_scatter_kernel<<<((size_t)E * 64 + 255) / 256, 256, 0, stream>>>(
            src, dst, dinv, hcur, agg, E);
        combine_kernel<<<(n64 + 255) / 256, 256, 0, stream>>>(
            agg, hcur, h0, dinv, hnxt, n64);
        float* t = hcur; hcur = hnxt; hnxt = t;
    }

    // log-softmax over rows of 64
    log_softmax_kernel<<<(N + 7) / 8, 256, 0, stream>>>(hcur, (float*)d_out, N);
}
